// GNN_29076928594164
// MI455X (gfx1250) — matrix operand order repaired and checked by reading, not compiled, not run on hardware
//
#include <hip/hip_runtime.h>
#include <hip/hip_bf16.h>

// GNN propagation (attention + fusion conv + ConvGRU) for MI455X / gfx1250.
// All GEMM-shaped math (attention einsums, 1x1 conv, implicit-GEMM 3x3 conv)
// runs on V_WMMA_F32_16X16X32_BF16 (wave32), f32 accumulate, with on-the-fly
// f32->bf16 conversion during the global->LDS tile staging.
// Block = 256 threads (8 waves, 4x2), macro tile 128(M) x 128(N), 32x64 per wave.

typedef __bf16 bf16;
typedef __attribute__((ext_vector_type(16))) __bf16 v16bf;
typedef __attribute__((ext_vector_type(8)))  __bf16 bf16x8;
typedef __attribute__((ext_vector_type(8)))  float  v8f;

#define HWN 1024   // H*W (also the fixed row stride of every B source and D dest)
#define CCH 512    // channels

__device__ __forceinline__ bf16 f2bf(float f) {
  unsigned u = __builtin_bit_cast(unsigned, f);
  unsigned r = u + 0x7fffu + ((u >> 16) & 1u);
  unsigned short h = (unsigned short)(r >> 16);
  return __builtin_bit_cast(bf16, h);
}

__device__ __forceinline__ unsigned pk2bf(float a, float b) {
  unsigned ua = __builtin_bit_cast(unsigned, a);
  ua = (ua + 0x7fffu + ((ua >> 16) & 1u)) >> 16;
  unsigned ub = __builtin_bit_cast(unsigned, b);
  ub = (ub + 0x7fffu + ((ub >> 16) & 1u)) >> 16;
  return ua | (ub << 16);
}

// D[M,N] = act( A * B + bias ), f32 in/out, bf16 WMMA inside.
// AMODE 0: A row-major [M,K], lda = row stride.
// AMODE 1: A stored [K,M] (k-major), lda = stride between k rows.
// BMODE 0: B[k,n] = (k<csplit ? B0[k] : B1[k-csplit]), rows n-contig w/ stride 1024
//          (channel-concat 1x1 conv / attention logits).
// BMODE 1: B[k,n] = B0[n*1024 + k]  (transposed source, att = Q*S^T).
// BMODE 2: implicit-GEMM 3x3 conv gather over 32x32 images, K = 2C*9, split at csplit.
// ACT 0: none, 1: sigmoid, 2: tanh.  HAS_BIAS: compile-time per-M bias add.
template <int AMODE, int BMODE, int ACT, int HAS_BIAS>
__global__ __launch_bounds__(256) void gemm_wmma(
    const float* __restrict__ A, long long strideA, int lda,
    const float* __restrict__ B0, const float* __restrict__ B1,
    long long strideB, int csplit,
    const float* __restrict__ bias,
    float* __restrict__ D, long long strideD,
    int K) {
  constexpr int LS = 48;  // LDS row stride in bf16 elems (96B: 16B aligned, conflict-friendly)
  __shared__ alignas(16) bf16 As[128 * LS];
  __shared__ alignas(16) bf16 Bs[128 * LS];

  const int tid  = threadIdx.x;
  const int mblk = blockIdx.y * 128;
  const int nblk = blockIdx.x * 128;
  const int b    = blockIdx.z;

  const int wv = tid >> 5, lane = tid & 31;
  const int wm = wv >> 1, wn = wv & 1;        // 4x2 wave grid, 32x64 per wave
  const int lrow = lane & 15, lhalf = lane >> 4;

  v8f acc[2][4] = {};

  for (int k0 = 0; k0 < K; k0 += 32) {
    // ---------------- stage A tile (128 x 32) into As[m][k] ----------------
    if (AMODE == 0) {
      const float* Ab = A + (long long)b * strideA;
#pragma unroll
      for (int j = 0; j < 4; ++j) {
        int idx = tid + j * 256;           // 1024 float4 groups
        int row = idx >> 3, c4 = idx & 7;
        const float* p = Ab + (long long)(mblk + row) * lda + k0 + c4 * 4;
        float4 v = *(const float4*)p;
        uint2 w;
        w.x = pk2bf(v.x, v.y);
        w.y = pk2bf(v.z, v.w);
        *(uint2*)&As[row * LS + c4 * 4] = w;
      }
    } else {  // A stored [K, M]
      const float* Ab = A + (long long)b * strideA;
#pragma unroll
      for (int j = 0; j < 16; ++j) {
        int idx = tid + j * 256;           // 4096 scalars, m fastest (coalesced)
        int kk = idx >> 7, mm = idx & 127;
        As[mm * LS + kk] = f2bf(Ab[(long long)(k0 + kk) * lda + mblk + mm]);
      }
    }
    // ---------------- stage B tile (32 x 128) into Bs[n][k] ----------------
    if (BMODE == 0) {
#pragma unroll
      for (int j = 0; j < 16; ++j) {
        int idx = tid + j * 256;           // 4096 scalars, n fastest (coalesced)
        int kk = idx >> 7, nn = idx & 127;
        int ch = k0 + kk;
        const float* src = (ch < csplit)
            ? (B0 + (long long)b * strideB + (long long)ch * HWN)
            : (B1 + (long long)b * strideB + (long long)(ch - csplit) * HWN);
        Bs[nn * LS + kk] = f2bf(src[nblk + nn]);
      }
    } else if (BMODE == 1) {  // source [N, K]
#pragma unroll
      for (int j = 0; j < 4; ++j) {
        int idx = tid + j * 256;           // 1024 float4 groups
        int c4 = idx & 7, nn = idx >> 3;   // nn: 0..127
        const float* p = B0 + (long long)b * strideB + (long long)(nblk + nn) * HWN + k0 + c4 * 4;
        float4 v = *(const float4*)p;
        uint2 w;
        w.x = pk2bf(v.x, v.y);
        w.y = pk2bf(v.z, v.w);
        *(uint2*)&Bs[nn * LS + c4 * 4] = w;
      }
    } else {  // BMODE == 2: implicit-GEMM 3x3, pad=1, 32x32 images
#pragma unroll
      for (int j = 0; j < 16; ++j) {
        int idx = tid + j * 256;
        int kk = idx >> 7, nn = idx & 127;
        int kg = k0 + kk;
        int ci = kg / 9;
        int t9 = kg - ci * 9;
        int r = t9 / 3 - 1;
        int s = t9 - (t9 / 3) * 3 - 1;
        int p  = nblk + nn;
        int hh = (p >> 5) + r, ww = (p & 31) + s;
        float v = 0.0f;
        if ((unsigned)hh < 32u && (unsigned)ww < 32u) {
          const float* src = (ci < csplit)
              ? (B0 + (long long)b * strideB + (long long)ci * HWN)
              : (B1 + (long long)b * strideB + (long long)(ci - csplit) * HWN);
          v = src[hh * 32 + ww];
        }
        Bs[nn * LS + kk] = f2bf(v);
      }
    }
    __syncthreads();

    // ---------------- WMMA fragments, contiguous k order (k = 16*half + e) in A and B ----------------
    union U { v16bf v; bf16x8 h[2]; };
    v16bf afr[2], bfr[4];
#pragma unroll
    for (int i = 0; i < 2; ++i) {
      const bf16* ap = &As[(wm * 32 + i * 16 + lrow) * LS];
      U ua;
      ua.h[0] = *(const bf16x8*)(ap + 16 * lhalf);       // e0..7  -> K = e + 16*half
      ua.h[1] = *(const bf16x8*)(ap + 8 + 16 * lhalf);   // e8..15 -> K = e + 16*half
      afr[i] = ua.v;
    }
#pragma unroll
    for (int jj = 0; jj < 4; ++jj) {
      const bf16* bp = &Bs[(wn * 64 + jj * 16 + lrow) * LS + 16 * lhalf];  // K = e + 16*half
      U ub;
      ub.h[0] = *(const bf16x8*)(bp);
      ub.h[1] = *(const bf16x8*)(bp + 8);
      bfr[jj] = ub.v;
    }
#pragma unroll
    for (int i = 0; i < 2; ++i)
#pragma unroll
      for (int jj = 0; jj < 4; ++jj)
        acc[i][jj] = __builtin_amdgcn_wmma_f32_16x16x32_bf16(
            false, afr[i], false, bfr[jj], (short)0, acc[i][jj], false, false);
    __syncthreads();
  }

  // ---------------- epilogue: bias + activation, f32 store ----------------
#pragma unroll
  for (int i = 0; i < 2; ++i) {
    const int mbase = mblk + wm * 32 + i * 16 + 8 * lhalf;  // 8 contiguous M rows per lane
    float bv[8];
    if (HAS_BIAS) {
      float4 b0 = *(const float4*)(bias + mbase);
      float4 b1 = *(const float4*)(bias + mbase + 4);
      bv[0] = b0.x; bv[1] = b0.y; bv[2] = b0.z; bv[3] = b0.w;
      bv[4] = b1.x; bv[5] = b1.y; bv[6] = b1.z; bv[7] = b1.w;
    }
#pragma unroll
    for (int jj = 0; jj < 4; ++jj) {
      float* dp = D + (long long)b * strideD + (long long)mbase * HWN +
                  nblk + wn * 64 + jj * 16 + lrow;
#pragma unroll
      for (int r = 0; r < 8; ++r) {
        float v = acc[i][jj][r];
        if (HAS_BIAS) v += bv[r];
        if (ACT == 1) v = 1.0f / (1.0f + __expf(-v));
        else if (ACT == 2) v = tanhf(v);
        dp[r * HWN] = v;   // compile-time stride -> immediate-offset stores
      }
    }
  }
}

// In-place softmax over rows of length 1024, one 256-thread block per row.
__global__ __launch_bounds__(256) void softmax_rows(float* __restrict__ X) {
  float* x = X + (long long)blockIdx.x * HWN;
  const int tid = threadIdx.x;
  const int wv = tid >> 5, lane = tid & 31;
  float4 d = *(float4*)(x + tid * 4);
  float m = fmaxf(fmaxf(d.x, d.y), fmaxf(d.z, d.w));
  for (int o = 16; o > 0; o >>= 1) m = fmaxf(m, __shfl_xor(m, o, 32));
  __shared__ float smax[8];
  __shared__ float ssum[8];
  if (lane == 0) smax[wv] = m;
  __syncthreads();
  float bm = smax[0];
#pragma unroll
  for (int i = 1; i < 8; ++i) bm = fmaxf(bm, smax[i]);
  d.x = __expf(d.x - bm); d.y = __expf(d.y - bm);
  d.z = __expf(d.z - bm); d.w = __expf(d.w - bm);
  float s = d.x + d.y + d.z + d.w;
  for (int o = 16; o > 0; o >>= 1) s += __shfl_xor(s, o, 32);
  if (lane == 0) ssum[wv] = s;
  __syncthreads();
  float bs = 0.0f;
#pragma unroll
  for (int i = 0; i < 8; ++i) bs += ssum[i];
  float inv = 1.0f / bs;
  d.x *= inv; d.y *= inv; d.z *= inv; d.w *= inv;
  *(float4*)(x + tid * 4) = d;
}

__global__ __launch_bounds__(256) void mul_ew(const float* __restrict__ a,
                                              const float* __restrict__ b,
                                              float* __restrict__ o, long long n) {
  long long i = (long long)blockIdx.x * blockDim.x + threadIdx.x;
  long long st = (long long)gridDim.x * blockDim.x;
  for (; i < n; i += st) o[i] = a[i] * b[i];
}

__global__ __launch_bounds__(256) void gru_update_ew(float* __restrict__ h,
                                                     const float* __restrict__ u,
                                                     const float* __restrict__ c,
                                                     long long n) {
  long long i = (long long)blockIdx.x * blockDim.x + threadIdx.x;
  long long st = (long long)gridDim.x * blockDim.x;
  for (; i < n; i += st) {
    float uu = u[i];
    h[i] = h[i] * (1.0f - uu) + c[i] * uu;
  }
}

extern "C" void kernel_launch(void* const* d_in, const int* in_sizes, int n_in,
                              void* d_out, int out_size, void* d_ws, size_t ws_size,
                              hipStream_t stream) {
  (void)in_sizes; (void)n_in; (void)out_size; (void)ws_size;
  const float* in1      = (const float*)d_in[0];
  const float* in2      = (const float*)d_in[1];
  const float* in3      = (const float*)d_in[2];
  const float* w_fusion = (const float*)d_in[3];
  const float* b_fusion = (const float*)d_in[4];
  const float* w_out    = (const float*)d_in[5];
  const float* b_out    = (const float*)d_in[6];
  const float* w_reset  = (const float*)d_in[7];
  const float* b_reset  = (const float*)d_in[8];
  const float* w_update = (const float*)d_in[9];
  const float* b_update = (const float*)d_in[10];
  const float* w_cand   = (const float*)d_in[11];
  const float* b_cand   = (const float*)d_in[12];
  float* out = (float*)d_out;

  const int Bc = 16;
  const long long S1 = (long long)Bc * CCH * HWN;  // floats per [B,C,H,W]
  const long long CS = (long long)CCH * HWN;       // per-batch activation stride
  const long long SA = (long long)HWN * HWN;       // per-batch attention-matrix stride

  float* ws  = (float*)d_ws;
  float* exb = ws;                 // state ex
  float* qb  = ws + S1;            // state q
  float* q1b = ws + 2 * S1;        // state q1
  float* cat = ws + 3 * S1;        // [B, 2C, HW] attention pair (2*S1)
  float* Aat = ws + 5 * S1;        // [B, 1024, 1024] logits/softmax (2*S1)
  float* a1  = ws + 7 * S1;
  float* a2  = ws + 8 * S1;
  float* a3  = ws + 9 * S1;
  float* rh  = ws + 10 * S1;
  float* cc  = ws + 11 * S1;
  float* rb  = Aat;                // GRU phase reuses attention scratch
  float* ub  = Aat + S1;

  hipMemcpyAsync(exb, in1, S1 * sizeof(float), hipMemcpyDeviceToDevice, stream);
  hipMemcpyAsync(qb,  in2, S1 * sizeof(float), hipMemcpyDeviceToDevice, stream);
  hipMemcpyAsync(q1b, in3, S1 * sizeof(float), hipMemcpyDeviceToDevice, stream);

  dim3 blk(256);

  // att(ef,qf): A[n,m]=sum_c ef[c,n]qf[c,m]; S=softmax_m; dst[c,n]=sum_m qf[c,m]S[n,m]
  auto attention = [&](const float* ef, const float* qf, float* dst) {
    gemm_wmma<1, 0, 0, 0><<<dim3(8, 8, Bc), blk, 0, stream>>>(
        ef, CS, HWN, qf, qf, CS, CCH, nullptr,
        Aat, SA, CCH);
    softmax_rows<<<dim3(Bc * HWN), blk, 0, stream>>>(Aat);
    gemm_wmma<0, 1, 0, 0><<<dim3(8, 4, Bc), blk, 0, stream>>>(
        qf, CS, HWN, Aat, Aat, SA, 0, nullptr,
        dst, 2 * CS, HWN);
  };

  auto fusion = [&](float* dst) {
    gemm_wmma<0, 2, 0, 1><<<dim3(8, 4, Bc), blk, 0, stream>>>(
        w_fusion, 0, 2 * CCH * 9, cat, cat + CS, 2 * CS, CCH, b_fusion,
        dst, CS, 2 * CCH * 9);
  };

  auto gru = [&](const float* x, float* h) {
    gemm_wmma<0, 0, 1, 1><<<dim3(8, 4, Bc), blk, 0, stream>>>(
        w_reset, 0, 2 * CCH, x, h, CS, CCH, b_reset,
        rb, CS, 2 * CCH);
    gemm_wmma<0, 0, 1, 1><<<dim3(8, 4, Bc), blk, 0, stream>>>(
        w_update, 0, 2 * CCH, x, h, CS, CCH, b_update,
        ub, CS, 2 * CCH);
    mul_ew<<<dim3(4096), blk, 0, stream>>>(rb, h, rh, S1);
    gemm_wmma<0, 0, 2, 1><<<dim3(8, 4, Bc), blk, 0, stream>>>(
        w_cand, 0, 2 * CCH, x, rh, CS, CCH, b_cand,
        cc, CS, 2 * CCH);
    gru_update_ew<<<dim3(4096), blk, 0, stream>>>(h, ub, cc, S1);
  };

  for (int round = 0; round < 5; ++round) {
    attention(exb, qb,  cat);
    attention(exb, q1b, cat + CS);
    fusion(a1);
    attention(qb, exb,  cat);
    attention(qb, q1b,  cat + CS);
    fusion(a2);
    attention(q1b, exb, cat);
    attention(q1b, qb,  cat + CS);
    fusion(a3);
    gru(a1, exb);
    gru(a2, qb);
    gru(a3, q1b);
  }

  const float* ins[3]  = {in1, in2, in3};
  float* states[3]     = {exb, qb, q1b};
  for (int i = 0; i < 3; ++i) {
    gemm_wmma<0, 2, 0, 1><<<dim3(8, 4, Bc), blk, 0, stream>>>(
        w_out, 0, 2 * CCH * 9, states[i], ins[i], CS, CCH, b_out,
        out + (long long)i * S1, CS, 2 * CCH * 9);
  }
}
